// BitNetLinear_89489938580129
// MI455X (gfx1250) — matrix operand order repaired and checked by reading, not compiled, not run on hardware
//
#include <hip/hip_runtime.h>
#include <hip/hip_bf16.h>

// ---------------------------------------------------------------------------
// BitNetLinear on gfx1250 (MI455X):
//   GEMM M=4096 (2*2048), N=4096, K=4096 via V_WMMA_F32_16X16X32_BF16,
//   with async global->LDS double-buffered staging (ASYNCcnt pipeline).
//   Ternary weights {-1,0,1} are exact in bf16; f32 accumulation.
//   out = acc * (wscale*iscale) + bias*iscale, acc = (x/iscale)_bf16 @ qw^T
// ---------------------------------------------------------------------------

typedef __attribute__((ext_vector_type(16))) __bf16 v16bf;
typedef __attribute__((ext_vector_type(8)))  float  v8f;

#define KDIM 4096
#define NDIM 4096

// GEMM tiling
#define BM 128
#define BN 128
#define BK 64                  // K per pipeline stage (2 WMMA K-steps)
#define SROW 144               // LDS row stride: 128B of k-data + 16B pad
#define STILE (128 * SROW)     // 18432 B per (matrix, buffer)
#define NT (KDIM / BK)         // 64 pipeline stages

__device__ __forceinline__ unsigned short f2bf(float f) {
    // round-to-nearest-even f32 -> bf16 (inputs here are finite)
    unsigned int u = __float_as_uint(f);
    u += 0x7FFFu + ((u >> 16) & 1u);
    return (unsigned short)(u >> 16);
}

// ---- pass 1: per-block partial sums of |v| ---------------------------------
__global__ __launch_bounds__(256)
void absum_partial_kernel(const float* __restrict__ p, int n4, float* __restrict__ out) {
    __shared__ float s[256];
    const float4* p4 = (const float4*)p;
    float acc = 0.f;
    int stride = gridDim.x * blockDim.x;
    for (int i = blockIdx.x * blockDim.x + threadIdx.x; i < n4; i += stride) {
        float4 v = p4[i];
        acc += fabsf(v.x) + fabsf(v.y) + fabsf(v.z) + fabsf(v.w);
    }
    s[threadIdx.x] = acc;
    __syncthreads();
    for (int o = 128; o > 0; o >>= 1) {
        if (threadIdx.x < (unsigned)o) s[threadIdx.x] += s[threadIdx.x + o];
        __syncthreads();
    }
    if (threadIdx.x == 0) out[blockIdx.x] = s[0];
}

// ---- pass 2: finalize scales (deterministic fixed-order tree) --------------
__global__ __launch_bounds__(256)
void finalize_scales_kernel(const float* __restrict__ pw, const float* __restrict__ px,
                            int nb, float* __restrict__ scales,
                            float wcount, float xcount) {
    __shared__ float s[256];
    __shared__ float wsum_sh;
    float a = 0.f;
    for (int i = threadIdx.x; i < nb; i += 256) a += pw[i];
    s[threadIdx.x] = a;
    __syncthreads();
    for (int o = 128; o > 0; o >>= 1) {
        if (threadIdx.x < (unsigned)o) s[threadIdx.x] += s[threadIdx.x + o];
        __syncthreads();
    }
    if (threadIdx.x == 0) wsum_sh = s[0];
    __syncthreads();
    a = 0.f;
    for (int i = threadIdx.x; i < nb; i += 256) a += px[i];
    s[threadIdx.x] = a;
    __syncthreads();
    for (int o = 128; o > 0; o >>= 1) {
        if (threadIdx.x < (unsigned)o) s[threadIdx.x] += s[threadIdx.x + o];
        __syncthreads();
    }
    if (threadIdx.x == 0) {
        float wscale = fmaxf(wsum_sh / wcount, 1e-8f);
        float iscale = fmaxf(s[0] / xcount, 1e-8f);
        scales[0] = wscale;            // weight scale
        scales[1] = iscale;            // input scale (bias multiplier)
        scales[2] = 1.0f / iscale;     // activation pre-multiplier
        scales[3] = wscale * iscale;   // GEMM epilogue multiplier
    }
}

// ---- ternary-quantize W to bf16 {-1,0,1}, row-major (OUT x IN) -------------
__global__ __launch_bounds__(256)
void quantize_w_kernel(const float* __restrict__ w, unsigned int* __restrict__ q,
                       int n4, const float* __restrict__ scales) {
    const float th = 0.5f * scales[0];
    const float4* w4 = (const float4*)w;
    uint2* q2 = (uint2*)q;
    int stride = gridDim.x * blockDim.x;
    for (int i = blockIdx.x * blockDim.x + threadIdx.x; i < n4; i += stride) {
        float4 v = w4[i];
        float q0 = (fabsf(v.x) > th) ? copysignf(1.f, v.x) : 0.f;
        float q1 = (fabsf(v.y) > th) ? copysignf(1.f, v.y) : 0.f;
        float q2f = (fabsf(v.z) > th) ? copysignf(1.f, v.z) : 0.f;
        float q3 = (fabsf(v.w) > th) ? copysignf(1.f, v.w) : 0.f;
        uint2 o;
        o.x = (unsigned)f2bf(q0)  | ((unsigned)f2bf(q1) << 16);
        o.y = (unsigned)f2bf(q2f) | ((unsigned)f2bf(q3) << 16);
        q2[i] = o;
    }
}

// ---- scale x by 1/input_scale, convert to bf16 -----------------------------
__global__ __launch_bounds__(256)
void scale_x_kernel(const float* __restrict__ x, unsigned int* __restrict__ xs,
                    int n4, const float* __restrict__ scales) {
    const float inv = scales[2];
    const float4* x4 = (const float4*)x;
    uint2* o2 = (uint2*)xs;
    int stride = gridDim.x * blockDim.x;
    for (int i = blockIdx.x * blockDim.x + threadIdx.x; i < n4; i += stride) {
        float4 v = x4[i];
        uint2 o;
        o.x = (unsigned)f2bf(v.x * inv) | ((unsigned)f2bf(v.y * inv) << 16);
        o.y = (unsigned)f2bf(v.z * inv) | ((unsigned)f2bf(v.w * inv) << 16);
        o2[i] = o;
    }
}

// ---- WMMA GEMM with async global->LDS double buffering ---------------------
// A = scaled x (bf16, M x K row-major), B = ternary W (bf16, N x K row-major:
// a column of the mathematical B = W^T is a contiguous row of W).
// Block tile 128x128, 8 waves (4Mx2N), wave tile 32x64 -> 16 WMMA per BK=64.
__global__ __launch_bounds__(256)
void bitnet_wmma_gemm(const unsigned short* __restrict__ Ar,
                      const unsigned short* __restrict__ Br,
                      const float* __restrict__ bias,
                      const float* __restrict__ scales,
                      float* __restrict__ out) {
    __shared__ __align__(16) unsigned char smem[2][2][STILE]; // [buf][A/B][..]

    const int t     = threadIdx.x;
    const int lane  = t & 31;
    const int wave  = t >> 5;
    const int waveM = wave >> 1;                 // 0..3
    const int waveN = wave & 1;                  // 0..1
    const int half  = lane >> 4;                 // WMMA lane group
    const int r     = lane & 15;                 // row/col within 16-tile

    const int m0 = blockIdx.y * BM + waveM * 32;
    const int n0 = blockIdx.x * BN + waveN * 64;

    // This thread's async-copy slice: row = t>>1, 64B half-row = t&1.
    const int grow  = t >> 1;
    const int ghalf = t & 1;
    const unsigned char* gA = (const unsigned char*)Ar
        + (size_t)(blockIdx.y * BM + grow) * (KDIM * 2) + ghalf * 64;
    const unsigned char* gB = (const unsigned char*)Br
        + (size_t)(blockIdx.x * BN + grow) * (KDIM * 2) + ghalf * 64;
    const unsigned ldsThr = (unsigned)(grow * SROW + ghalf * 64);
    const unsigned sbase  = (unsigned)(unsigned long long)(void*)&smem[0][0][0];

    v8f acc[2][4] = {};

    // Issue one BK-stage: 8x global_load_async_to_lds_b128 per thread.
    // ISA GV mode adds INST_OFFSET to BOTH the LDS and global addresses,
    // so one address pair + offsets copies a contiguous 64B run.
    auto issue = [&](int kt, int buf) {
        const unsigned char* ga = gA + (size_t)kt * (BK * 2);
        const unsigned char* gb = gB + (size_t)kt * (BK * 2);
        unsigned la = sbase + (unsigned)(buf * (2 * STILE)) + ldsThr;
        unsigned lb = la + STILE;
        asm volatile(
            "global_load_async_to_lds_b128 %0, %2, off\n\t"
            "global_load_async_to_lds_b128 %0, %2, off offset:16\n\t"
            "global_load_async_to_lds_b128 %0, %2, off offset:32\n\t"
            "global_load_async_to_lds_b128 %0, %2, off offset:48\n\t"
            "global_load_async_to_lds_b128 %1, %3, off\n\t"
            "global_load_async_to_lds_b128 %1, %3, off offset:16\n\t"
            "global_load_async_to_lds_b128 %1, %3, off offset:32\n\t"
            "global_load_async_to_lds_b128 %1, %3, off offset:48"
            :: "v"(la), "v"(lb), "v"(ga), "v"(gb) : "memory");
    };

    issue(0, 0);
    for (int kt = 0; kt < NT; ++kt) {
        const int buf = kt & 1;
        if (kt + 1 < NT) {
            issue(kt + 1, buf ^ 1);                    // overlap next stage
            if (kt + 4 < NT) {                         // keep HBM->L2 warm
                __builtin_prefetch(gA + (size_t)(kt + 4) * (BK * 2), 0, 1);
                __builtin_prefetch(gB + (size_t)(kt + 4) * (BK * 2), 0, 1);
            }
            // async loads complete in order: <=8 outstanding => current done
            asm volatile("s_wait_asynccnt 0x8" ::: "memory");
        } else {
            asm volatile("s_wait_asynccnt 0x0" ::: "memory");
        }
        __syncthreads();

        const unsigned char* sA = &smem[buf][0][0];
        const unsigned char* sB = &smem[buf][1][0];
#pragma unroll
        for (int ks = 0; ks < 2; ++ks) {               // two 32-wide K-steps
            union { v16bf v; uint4 u[2]; } af[2], bfr[4];
#pragma unroll
            for (int i = 0; i < 2; ++i) {
                // A frag (ISA §7.12.2): lane group h: K bytes [ks*64+h*16)
                // in v[0:3], +32B in v[4:7], row M = r.
                const unsigned char* p =
                    sA + (waveM * 32 + i * 16 + r) * SROW + ks * 64 + half * 16;
                af[i].u[0] = *(const uint4*)p;
                af[i].u[1] = *(const uint4*)(p + 32);
            }
#pragma unroll
            for (int j = 0; j < 4; ++j) {
                // B frag: lane group h: K bytes [h*16) in v[0:3], +32B in v[4:7],
                // column N = r  ==> run of W's row.
                const unsigned char* p =
                    sB + (waveN * 64 + j * 16 + r) * SROW + ks * 64 + half * 16;
                bfr[j].u[0] = *(const uint4*)p;
                bfr[j].u[1] = *(const uint4*)(p + 32);
            }
#pragma unroll
            for (int i = 0; i < 2; ++i)
#pragma unroll
                for (int j = 0; j < 4; ++j)
                    acc[i][j] = __builtin_amdgcn_wmma_f32_16x16x32_bf16(
                        /*neg_a=*/false, af[i].v, /*neg_b=*/false, bfr[j].v,
                        /*c_mod=*/(short)0, acc[i][j],
                        /*reuse_a=*/false, /*reuse_b=*/false);
        }
        __syncthreads();                               // buf free for rewrite
    }

    const float outmul = scales[3];
    const float bmul   = scales[1];
#pragma unroll
    for (int i = 0; i < 2; ++i) {
#pragma unroll
        for (int j = 0; j < 4; ++j) {
            const int col = n0 + j * 16 + r;
            const float bb = bias[col] * bmul;
#pragma unroll
            for (int v = 0; v < 8; ++v) {
                // C/D layout: VGPR v, lanes 0-15 -> M=v, lanes 16-31 -> M=v+8
                const int row = m0 + i * 16 + v + half * 8;
                out[(size_t)row * NDIM + col] = acc[i][j][v] * outmul + bb;
            }
        }
    }
}

extern "C" void kernel_launch(void* const* d_in, const int* in_sizes, int n_in,
                              void* d_out, int out_size, void* d_ws, size_t ws_size,
                              hipStream_t stream) {
    const float* x    = (const float*)d_in[0];   // (2,2048,4096) f32
    const float* w    = (const float*)d_in[1];   // (4096,4096) f32
    const float* bias = (const float*)d_in[2];   // (4096,) f32
    float* out = (float*)d_out;

    const int nx = in_sizes[0];                  // 16777216
    const int nw = in_sizes[1];                  // 16777216
    const int M  = nx / KDIM;                    // 4096

    // workspace layout
    char* ws = (char*)d_ws;
    float* scales = (float*)ws;                            // 4 floats
    float* pw     = (float*)(ws + 256);                    // 1024 partials
    float* px     = (float*)(ws + 256 + 4096);             // 1024 partials
    unsigned short* qw = (unsigned short*)(ws + 65536);                      // N*K bf16
    unsigned short* xs = (unsigned short*)(ws + 65536 + (size_t)NDIM * KDIM * 2); // M*K bf16

    // 1) deterministic two-pass reductions for mean(|W|), mean(|x|)
    absum_partial_kernel<<<1024, 256, 0, stream>>>(w, nw >> 2, pw);
    absum_partial_kernel<<<1024, 256, 0, stream>>>(x, nx >> 2, px);
    finalize_scales_kernel<<<1, 256, 0, stream>>>(pw, px, 1024, scales,
                                                  (float)nw, (float)nx);

    // 2) quantize / scale to bf16
    quantize_w_kernel<<<4096, 256, 0, stream>>>(w, (unsigned int*)qw, nw >> 2, scales);
    scale_x_kernel<<<4096, 256, 0, stream>>>(x, (unsigned int*)xs, nx >> 2, scales);

    // 3) WMMA GEMM + fused epilogue (async-LDS double-buffered pipeline)
    dim3 grid(NDIM / BN, M / BM);
    bitnet_wmma_gemm<<<grid, 256, 0, stream>>>(xs, qw, bias, scales, out);
}
